// Net_16097537425785
// MI455X (gfx1250) — compile-verified, matrix operand order checked by reading, not run on hardware
//
#include <hip/hip_runtime.h>
#include <hip/hip_bf16.h>
#include <stdint.h>

// ---------------- CDNA5 WMMA types & fragment helpers ----------------
typedef __attribute__((ext_vector_type(16))) __bf16 v16bf;
typedef __attribute__((ext_vector_type(8)))  float  v8f;
typedef __attribute__((ext_vector_type(4)))  float  v4f;

union Frag16 { v16bf v; uint4 q[2]; };
union B4 { uint16_t h[4]; uint64_t u; };

__device__ __forceinline__ uint16_t f2bf(float f) {
    uint32_t u = __float_as_uint(f);
    u = u + 0x7fffu + ((u >> 16) & 1u);      // round-to-nearest-even
    return (uint16_t)(u >> 16);
}
__device__ __forceinline__ float bf2f(uint16_t h) {
    return __uint_as_float(((uint32_t)h) << 16);
}
__device__ __forceinline__ float sigmoidf_(float x) {
    return 1.0f / (1.0f + __expf(-x));
}

// A-fragment (16x32 bf16, M x K) from row-major bf16 in LDS.
// ISA layout: lanes 0-15: M=lane, K=0..7 (elems 0..7) and K=16..23 (elems 8..15)
//             lanes16-31: M=lane-16, K=8..15 and K=24..31  -> two contiguous b128 loads.
__device__ __forceinline__ Frag16 load_a(const uint16_t* base, int stride, int k0, int lane) {
    const uint16_t* p = base + (size_t)(lane & 15) * stride + k0 + ((lane >> 4) << 3);
    Frag16 f;
    f.q[0] = *reinterpret_cast<const uint4*>(p);
    f.q[1] = *reinterpret_cast<const uint4*>(p + 16);
    return f;
}

// B-fragment (32x16 bf16, K x N), pre-packed in global memory: per (nt,kt,lane)
// 8 contiguous dwords. Two contiguous b128 loads per lane (L2-resident; compiler
// may hoist loop-invariant fragments into VGPRs).
__device__ __forceinline__ Frag16 load_b(const uint32_t* pw, int KT, int kt, int nt, int lane) {
    const uint4* p = reinterpret_cast<const uint4*>(pw) + ((size_t)((nt * KT + kt) * 32 + lane)) * 2;
    Frag16 f;
    f.q[0] = p[0];
    f.q[1] = p[1];
    return f;
}

__device__ __forceinline__ v8f wmma_bf16(const Frag16& a, const Frag16& b, v8f c) {
    return __builtin_amdgcn_wmma_f32_16x16x32_bf16(false, a.v, false, b.v, (short)0, c, false, false);
}

// Software-pipelined GEMM chain: A from LDS (double-buffered so the next
// ds_load_b128 pair issues before the current WMMA), B from pre-packed global.
template <int KT>
__device__ __forceinline__ v8f gemm_chain(const uint16_t* abase, int stride,
                                          const uint32_t* pw, int nt, int lane) {
    v8f acc = {0.f,0.f,0.f,0.f,0.f,0.f,0.f,0.f};
    Frag16 a_cur = load_a(abase, stride, 0, lane);
#pragma unroll
    for (int kt = 0; kt < KT - 1; ++kt) {
        Frag16 a_nxt = load_a(abase, stride, (kt + 1) * 32, lane);   // prefetch k+1
        Frag16 b = load_b(pw, KT, kt, nt, lane);
        acc = wmma_bf16(a_cur, b, acc);
        a_cur = a_nxt;
    }
    Frag16 b = load_b(pw, KT, KT - 1, nt, lane);
    return wmma_bf16(a_cur, b, acc);
}

// K index inside a 32-wide k-tile for fragment element e, lane l (A and B share it).
__device__ __forceinline__ int kmap(int e, int lane) {
    return ((e >> 3) << 4) + ((lane >> 4) << 3) + (e & 7);
}

// ---------------- constants ----------------
static constexpr int Bn = 64, Tn = 128, In = 64, Hn = 256, Mn = 64, FFn = 512;
static constexpr int NTH = 512;   // 16 waves per scan workgroup

// ---------------- weight packing kernels (f32 -> bf16 WMMA B-fragments) ----------------
__global__ void pack_w(const float* __restrict__ W, int KT, int NT, int K, int N,
                       uint32_t* __restrict__ dst) {
    int tid = blockIdx.x * blockDim.x + threadIdx.x;
    int total = KT * NT * 32;
    if (tid >= total) return;
    int lane = tid & 31, f = tid >> 5;
    int kt = f % KT, nt = f / KT;
    int n = nt * 16 + (lane & 15);
    uint32_t* o = dst + (size_t)tid * 8;
#pragma unroll
    for (int d = 0; d < 8; ++d) {
        uint32_t pair = 0;
#pragma unroll
        for (int e2 = 0; e2 < 2; ++e2) {
            int e = d * 2 + e2;
            int k = kt * 32 + kmap(e, lane);
            float v = (k < K && n < N) ? W[(size_t)k * N + n] : 0.0f;
            pair |= (uint32_t)f2bf(v) << (16 * e2);
        }
        o[d] = pair;
    }
}

// Pack [Wxh(64xH); Wrh(HxH); Whh(HxH)] -> 576x256 combined B fragments.
__global__ void pack_comb(const float* __restrict__ Wxh, const float* __restrict__ Wrh,
                          const float* __restrict__ Whh, uint32_t* __restrict__ dst) {
    const int KT = 18, NT = 16, N = Hn;
    int tid = blockIdx.x * blockDim.x + threadIdx.x;
    int total = KT * NT * 32;
    if (tid >= total) return;
    int lane = tid & 31, f = tid >> 5;
    int kt = f % KT, nt = f / KT;
    int n = nt * 16 + (lane & 15);
    uint32_t* o = dst + (size_t)tid * 8;
#pragma unroll
    for (int d = 0; d < 8; ++d) {
        uint32_t pair = 0;
#pragma unroll
        for (int e2 = 0; e2 < 2; ++e2) {
            int e = d * 2 + e2;
            int k = kt * 32 + kmap(e, lane);
            float v;
            if (k < 64)       v = Wxh[(size_t)k * N + n];
            else if (k < 320) v = Wrh[(size_t)(k - 64) * N + n];
            else              v = Whh[(size_t)(k - 320) * N + n];
            pair |= (uint32_t)f2bf(v) << (16 * e2);
        }
        o[d] = pair;
    }
}

// ---------------- persistent recurrent scan: one WG per 16-batch tile ----------------
__launch_bounds__(NTH, 1)
__global__ void rnn_scan(const float* __restrict__ x,           // [64][128][64]
                         const float* __restrict__ Wwg, const float* __restrict__ bwg,
                         const float* __restrict__ Wrg, const float* __restrict__ brg,
                         const float* __restrict__ bc,  const float* __restrict__ bwp,
                         const float* __restrict__ brp, const float* __restrict__ bh,
                         const float* __restrict__ wa,  const float* __restrict__ wb,
                         const uint32_t* __restrict__ pWc,   const uint32_t* __restrict__ pWwp,
                         const uint32_t* __restrict__ pWrp,  const uint32_t* __restrict__ pWcomb,
                         float* __restrict__ memg,              // [64][64][256] f32 (L2-resident)
                         uint16_t* __restrict__ hid)            // [64][128][256] bf16
{
    __shared__ uint16_t s_oh[16][320];   // [o_t | h]  bf16 (A for GEMM1)
    __shared__ uint16_t s_ch[16][512];   // [c   | h]  bf16 (A for GEMM2/3)
    __shared__ uint16_t s_z [16][576];   // [o | r | h] bf16 (A for GEMM4)
    __shared__ float s_p1[16][64], s_p2[16][64];       // alpha / ar (in-place softmax)
    __shared__ float s_gw[16], s_go[16];
    __shared__ float s_Wwg[320], s_Wrg[320];
    __shared__ float s_bc[256], s_bh[256], s_bwp[64], s_brp[64];
    __shared__ float s_sc[4];            // wa, wb, bwg, brg

    const int tid  = threadIdx.x;
    const int wave = tid >> 5, lane = tid & 31;
    const int btile = blockIdx.x;        // 0..3 (16 batch rows each)

    // one-time preload of small weights/biases + zero h state
    for (int i = tid; i < 320; i += NTH) { s_Wwg[i] = Wwg[i]; s_Wrg[i] = Wrg[i]; }
    for (int i = tid; i < 256; i += NTH) { s_bc[i] = bc[i]; s_bh[i] = bh[i]; }
    for (int i = tid; i < 64;  i += NTH) { s_bwp[i] = bwp[i]; s_brp[i] = brp[i]; }
    if (tid == 0) { s_sc[0] = wa[0]; s_sc[1] = wb[0]; s_sc[2] = bwg[0]; s_sc[3] = brg[0]; }
    for (int i = tid; i < 16 * 256; i += NTH) {
        int b = i >> 8, n = i & 255;
        s_oh[b][64 + n] = 0; s_ch[b][256 + n] = 0; s_z[b][320 + n] = 0;
    }
    __syncthreads();

    for (int t = 0; t < Tn; ++t) {
        // ---- phase 1: stream in o_t (vectorized) + prefetch next step ----
        if (tid < 16 * 16) {             // 256 threads, one float4 each
            int b = tid >> 4, hc = (tid & 15) * 4;
            const float* xp = &x[((size_t)(btile * 16 + b) * Tn + t) * In + hc];
            v4f v = *reinterpret_cast<const v4f*>(xp);
            if (t + 1 < Tn) __builtin_prefetch(xp + In, 0, 0);   // global_prefetch_b8
            B4 p;
#pragma unroll
            for (int j = 0; j < 4; ++j) p.h[j] = f2bf(v[j]);
            *reinterpret_cast<uint64_t*>(&s_oh[b][hc]) = p.u;
            *reinterpret_cast<uint64_t*>(&s_z[b][hc])  = p.u;
        }
        __syncthreads();

        // ---- phase 2: c = sigmoid(oh @ Wc + bc)  [16x320]@[320x256], 16 waves x 1 n-tile
        {
            const int nt = wave;
            v8f acc = gemm_chain<10>(&s_oh[0][0], 320, pWc, nt, lane);
            int n = nt * 16 + (lane & 15);
            int mb = (lane >> 4) << 3;
#pragma unroll
            for (int r = 0; r < 8; ++r)
                s_ch[mb + r][n] = f2bf(sigmoidf_(acc[r] + s_bc[n]));
        }
        __syncthreads();

        // ---- phase 3: p1 = wa*tanh(ch@Wwp+bwp), p2 = wb*tanh(ch@Wrp+brp)  (waves 0..7)
        if (wave < 8) {
            const int which = wave >> 2, nt = wave & 3;
            const uint32_t* pw = which ? pWrp : pWwp;
            const float* bias  = which ? s_brp : s_bwp;
            const float scale  = which ? s_sc[1] : s_sc[0];
            v8f acc = gemm_chain<16>(&s_ch[0][0], 512, pw, nt, lane);
            int n = nt * 16 + (lane & 15);
            int mb = (lane >> 4) << 3;
#pragma unroll
            for (int r = 0; r < 8; ++r) {
                float v = scale * tanhf(acc[r] + bias[n]);
                if (which) s_p2[mb + r][n] = v; else s_p1[mb + r][n] = v;
            }
        }
        __syncthreads();

        // ---- phase 4: scalar gates + softmaxes (64 threads, one row each) ----
        if (tid < 64) {
            int job = tid >> 4, b = tid & 15;
            if (job < 2) {                         // gw / go sigmoid dot over oh
                const float* wv = job ? s_Wrg : s_Wwg;
                float acc = job ? s_sc[3] : s_sc[2];
                for (int k = 0; k < 320; ++k) acc += bf2f(s_oh[b][k]) * wv[k];
                if (job) s_go[b] = sigmoidf_(acc); else s_gw[b] = sigmoidf_(acc);
            } else {                               // softmax over M=64, in place
                float* p = (job == 2) ? &s_p1[b][0] : &s_p2[b][0];
                float mx = -1e30f;
                for (int m = 0; m < 64; ++m) mx = fmaxf(mx, p[m]);
                float sum = 0.f;
                for (int m = 0; m < 64; ++m) { float e = __expf(p[m] - mx); p[m] = e; sum += e; }
                float inv = 1.0f / sum;
                for (int m = 0; m < 64; ++m) p[m] *= inv;
            }
        }
        __syncthreads();

        // ---- phase 5: fused mem read/update + r = go * (ar @ mem), float4-wide ----
        // Each (b, hc..hc+3) column group is owned by one thread for the whole scan,
        // so no synchronization is ever needed on memg (stays L2-resident).
        for (int idx = tid; idx < 16 * 64; idx += NTH) {     // 1024 groups, 2 per thread
            int b = idx >> 6, hc = (idx & 63) * 4;
            v4f* mp = reinterpret_cast<v4f*>(memg + ((size_t)(btile * 16 + b) * Mn) * Hn + hc);
            B4 c4; c4.u = *reinterpret_cast<const uint64_t*>(&s_ch[b][hc]);
            v4f cv;
#pragma unroll
            for (int j = 0; j < 4; ++j) cv[j] = bf2f(c4.h[j]);
            const float gwb = s_gw[b];
            v4f acc = {0.f, 0.f, 0.f, 0.f};
            if (t == 0) {                                     // mem starts at zero
#pragma unroll 4
                for (int m = 0; m < Mn; ++m)
                    mp[(size_t)m * (Hn / 4)] = (gwb * s_p1[b][m]) * cv;
            } else {
#pragma unroll 4
                for (int m = 0; m < Mn; ++m) {
                    v4f mv = mp[(size_t)m * (Hn / 4)];
                    float al = s_p1[b][m];
                    acc += s_p2[b][m] * mv;
                    mp[(size_t)m * (Hn / 4)] = (gwb * al) * cv + (1.0f - al) * mv;
                }
            }
            const float gob = s_go[b];
            B4 r4;
#pragma unroll
            for (int j = 0; j < 4; ++j) r4.h[j] = f2bf(acc[j] * gob);
            *reinterpret_cast<uint64_t*>(&s_z[b][64 + hc]) = r4.u;
        }
        __syncthreads();

        // ---- phase 6: h_new = sigmoid([o|r|h] @ [Wxh;Wrh;Whh] + bh)  [16x576]@[576x256]
        {
            const int nt = wave;
            v8f acc = gemm_chain<18>(&s_z[0][0], 576, pWcomb, nt, lane);
            int n = nt * 16 + (lane & 15);
            int mb = (lane >> 4) << 3;
#pragma unroll
            for (int r = 0; r < 8; ++r) {
                int m = mb + r;
                uint16_t hv = f2bf(sigmoidf_(acc[r] + s_bh[n]));
                s_oh[m][64 + n] = hv;           // next GEMM1 input
                s_ch[m][256 + n] = hv;          // next GEMM2/3 input
                s_z[m][320 + n] = hv;           // next GEMM4 input
                hid[((size_t)(btile * 16 + m) * Tn + t) * Hn + n] = hv;
            }
        }
        __syncthreads();
    }
}

// ---------------- decode: sigmoid(hid @ Wd1 + bd1) @ Wd2 + bd2 ----------------
__launch_bounds__(256)
__global__ void decode_kernel(const uint16_t* __restrict__ hid, const uint32_t* __restrict__ pWd1,
                              const float* __restrict__ bd1, const float* __restrict__ Wd2,
                              const float* __restrict__ bd2, float* __restrict__ out) {
    __shared__ float s_dec[16][512];
    __shared__ float s_part[16][16];
    __shared__ float s_w2[512];
    const int tid = threadIdx.x, wave = tid >> 5, lane = tid & 31;
    const int row0 = blockIdx.x * 16;              // 512 blocks over 8192 rows

    for (int i = tid; i < 512; i += 256) s_w2[i] = Wd2[i];

    const uint16_t* arow = hid + (size_t)(row0 + (lane & 15)) * Hn + ((lane >> 4) << 3);

#pragma unroll
    for (int j = 0; j < 4; ++j) {                  // 8 waves x 4 n-tiles = 32 tiles (N=512)
        const int nt = wave * 4 + j;
        v8f acc = {0.f,0.f,0.f,0.f,0.f,0.f,0.f,0.f};
        Frag16 a_cur;
        a_cur.q[0] = *reinterpret_cast<const uint4*>(arow);
        a_cur.q[1] = *reinterpret_cast<const uint4*>(arow + 16);
#pragma unroll
        for (int kt = 0; kt < 8; ++kt) {           // K = 256, A double-buffered
            Frag16 a_nxt;
            if (kt < 7) {
                a_nxt.q[0] = *reinterpret_cast<const uint4*>(arow + (kt + 1) * 32);
                a_nxt.q[1] = *reinterpret_cast<const uint4*>(arow + (kt + 1) * 32 + 16);
            }
            Frag16 b = load_b(pWd1, 8, kt, nt, lane);
            acc = wmma_bf16(a_cur, b, acc);
            if (kt < 7) a_cur = a_nxt;
        }
        int n = nt * 16 + (lane & 15);
        int mb = (lane >> 4) << 3;
#pragma unroll
        for (int r = 0; r < 8; ++r)
            s_dec[mb + r][n] = sigmoidf_(acc[r] + bd1[n]);
    }
    __syncthreads();

    // out = s_dec @ Wd2 + bd2 : 256 threads -> (row, 32-wide chunk) partials
    {
        int b = tid >> 4, c = tid & 15;
        float acc = 0.f;
        for (int k = c * 32; k < c * 32 + 32; ++k) acc += s_dec[b][k] * s_w2[k];
        s_part[b][c] = acc;
    }
    __syncthreads();
    if (tid < 16) {
        float acc = bd2[0];
        for (int c = 0; c < 16; ++c) acc += s_part[tid][c];
        out[row0 + tid] = acc;
    }
}

// ---------------- host launcher ----------------
extern "C" void kernel_launch(void* const* d_in, const int* in_sizes, int n_in,
                              void* d_out, int out_size, void* d_ws, size_t ws_size,
                              hipStream_t stream) {
    const float* x   = (const float*)d_in[0];
    const float* Wc  = (const float*)d_in[2];
    const float* bc  = (const float*)d_in[3];
    const float* Wwg = (const float*)d_in[4];
    const float* bwg = (const float*)d_in[5];
    const float* Wwp = (const float*)d_in[6];
    const float* bwp = (const float*)d_in[7];
    const float* Wrg = (const float*)d_in[8];
    const float* brg = (const float*)d_in[9];
    const float* Wrp = (const float*)d_in[10];
    const float* brp = (const float*)d_in[11];
    const float* Wd1 = (const float*)d_in[12];
    const float* bd1 = (const float*)d_in[13];
    const float* Wd2 = (const float*)d_in[14];
    const float* bd2 = (const float*)d_in[15];
    const float* Wxh = (const float*)d_in[16];
    const float* Wrh = (const float*)d_in[17];
    const float* Whh = (const float*)d_in[18];
    const float* wa  = (const float*)d_in[19];
    const float* wb  = (const float*)d_in[20];
    const float* bh  = (const float*)d_in[21];

    uint8_t* ws = (uint8_t*)d_ws;
    uint32_t* pWc    = (uint32_t*)(ws + 0);        // 320*256*2  = 163840
    uint32_t* pWwp   = (uint32_t*)(ws + 163840);   // 512*64*2   =  65536
    uint32_t* pWrp   = (uint32_t*)(ws + 229376);   //            =  65536
    uint32_t* pWcomb = (uint32_t*)(ws + 294912);   // 576*256*2  = 294912
    uint32_t* pWd1   = (uint32_t*)(ws + 589824);   // 256*512*2  = 262144
    float*    memg   = (float*)   (ws + 851968);   // 64*64*256*4 = 4 MB
    uint16_t* hid    = (uint16_t*)(ws + 5046272);  // 64*128*256*2 = 4 MB
    float*    out    = (float*)d_out;

    // pack weights into WMMA B-fragment order (bf16)
    pack_w<<<(10 * 16 * 32 + 255) / 256, 256, 0, stream>>>(Wc, 10, 16, 320, 256, pWc);
    pack_w<<<(16 * 4 * 32 + 255) / 256, 256, 0, stream>>>(Wwp, 16, 4, 512, 64, pWwp);
    pack_w<<<(16 * 4 * 32 + 255) / 256, 256, 0, stream>>>(Wrp, 16, 4, 512, 64, pWrp);
    pack_w<<<(8 * 32 * 32 + 255) / 256, 256, 0, stream>>>(Wd1, 8, 32, 256, 512, pWd1);
    pack_comb<<<(18 * 16 * 32 + 255) / 256, 256, 0, stream>>>(Wxh, Wrh, Whh, pWcomb);

    // persistent recurrent scan: 4 independent batch tiles, no global sync
    rnn_scan<<<4, NTH, 0, stream>>>(x, Wwg, bwg, Wrg, brg, bc, bwp, brp, bh, wa, wb,
                                    pWc, pWwp, pWrp, pWcomb, memg, hid);

    // wide parallel decode
    decode_kernel<<<512, 256, 0, stream>>>(hid, pWd1, bd1, Wd2, bd2, out);
}
